// CAModel_64888365908141
// MI455X (gfx1250) — matrix operand order repaired and checked by reading, not compiled, not run on hardware
//
#include <hip/hip_runtime.h>
#include <stdint.h>

typedef __attribute__((ext_vector_type(16))) _Float16 v16h;
typedef __attribute__((ext_vector_type(8)))  _Float16 v8h;
typedef __attribute__((ext_vector_type(8)))  float    v8f;
typedef __attribute__((ext_vector_type(4)))  float    v4f;

#define NCH   16
#define HH    64
#define WWD   64
#define NPIX  (HH * WWD)          // 4096 pixels per image
#define NTHREADS 256              // 8 waves (wave32)
#define NWAVES   8
#define TILES_PER_WAVE 32         // 256 16-pixel tiles / 8 waves
#define HROW  144                 // halfs per pixel row in h-scratch (128 + 16 pad for banks)

struct SMem {
  _Float16 sA[NCH * NPIX];                      // 128 KB state (read side)
  _Float16 sB[NCH * NPIX];                      // 128 KB state (write side / "out", pre-clip)
  alignas(32) _Float16 hscr[NWAVES][16][HROW];  // 36 KB layer1->layer2 reshuffle scratch
};

__device__ __forceinline__ uint32_t rotl32(uint32_t v, int n) { return (v << n) | (v >> (32 - n)); }

// JAX threefry2x32, 20 rounds (matches jax._src.prng.threefry2x32)
__device__ __forceinline__ void tf2x32(uint32_t k0, uint32_t k1, uint32_t x0, uint32_t x1,
                                       uint32_t& o0, uint32_t& o1) {
  const uint32_t ks2 = k0 ^ k1 ^ 0x1BD11BDAu;
  x0 += k0; x1 += k1;
#define TFR(rot) { x0 += x1; x1 = rotl32(x1, rot); x1 ^= x0; }
  TFR(13) TFR(15) TFR(26) TFR(6)  x0 += k1;  x1 += ks2 + 1u;
  TFR(17) TFR(29) TFR(16) TFR(24) x0 += ks2; x1 += k0  + 2u;
  TFR(13) TFR(15) TFR(26) TFR(6)  x0 += k0;  x1 += k1  + 3u;
  TFR(17) TFR(29) TFR(16) TFR(24) x0 += k1;  x1 += ks2 + 4u;
  TFR(13) TFR(15) TFR(26) TFR(6)  x0 += ks2; x1 += k0  + 5u;
#undef TFR
  o0 = x0; o1 = x1;
}

// random_bits(key, count j) with total count 2*half: jax splits counts into halves
__device__ __forceinline__ uint32_t tf_bits(uint32_t k0, uint32_t k1, uint32_t j, uint32_t half) {
  uint32_t o0, o1;
  if (j < half) { tf2x32(k0, k1, j, j + half, o0, o1); return o0; }
  tf2x32(k0, k1, j - half, j, o0, o1); return o1;
}

// zero-padded read of state channel plane (for 'SAME' conv padding)
__device__ __forceinline__ float ldz(const _Float16* s, int c, int r, int col) {
  if ((unsigned)r >= (unsigned)HH || (unsigned)col >= (unsigned)WWD) return 0.0f;
  return (float)s[c * NPIX + r * WWD + col];
}

__global__ __launch_bounds__(NTHREADS, 1)
void nca_persistent_kernel(const float* __restrict__ x0,
                           const float* __restrict__ W1, const float* __restrict__ b1,
                           const float* __restrict__ W2, const float* __restrict__ b2,
                           const int* __restrict__ n_steps_p,
                           float* __restrict__ out) {
  __shared__ SMem sm;
  const int tid  = threadIdx.x;
  const int b    = blockIdx.x;
  const int lane = tid & 31;
  const int wave = tid >> 5;
  const int lrow = lane & 15;   // M row (A/weights) or pixel-in-tile (B/C/D)
  const int lgrp = lane >> 4;   // lane group
  const int ns   = *n_steps_p;

  // ---- Stage weights once into per-lane WMMA A-matrix register layout (f16) ----
  // A element e -> K = lgrp*16+e; row M = lrow
  v16h A1[16];  // layer1: 8 M-tiles x 2 K-tiles (K padded 48 -> 64)
#pragma unroll
  for (int m = 0; m < 8; ++m)
#pragma unroll
    for (int kt = 0; kt < 2; ++kt) {
      v16h a;
#pragma unroll
      for (int e = 0; e < 16; ++e) {
        int K  = lgrp * 16 + e;
        int ch = kt * 32 + K;
        float w = (ch < 48) ? W1[(m * 16 + lrow) * 48 + ch] : 0.0f;
        a[e] = (_Float16)w;
      }
      A1[m * 2 + kt] = a;
    }
  v16h A2[4];   // layer2: 1 M-tile x 4 K-tiles (K = 128)
#pragma unroll
  for (int j = 0; j < 4; ++j) {
    v16h a;
#pragma unroll
    for (int e = 0; e < 16; ++e) {
      int K = lgrp * 16 + e;
      a[e] = (_Float16)W2[lrow * 128 + j * 32 + K];
    }
    A2[j] = a;
  }

  // ---- Biases in registers, pre-swizzled to the C/D lane layout (loop-invariant) ----
  v8f bias1[8];
#pragma unroll
  for (int m = 0; m < 8; ++m)
#pragma unroll
    for (int r = 0; r < 8; ++r) bias1[m][r] = b1[m * 16 + lgrp * 8 + r];
  v8f bias2;
#pragma unroll
  for (int r = 0; r < 8; ++r) bias2[r] = b2[lgrp * 8 + r];

  // ---- Load initial state (f32 global -> f16 LDS), layout identical to NCHW ----
  {
    const float* src = x0 + (size_t)b * NCH * NPIX;
#pragma unroll 1
    for (int i0 = tid * 256; i0 < tid * 256 + 256; i0 += 8) {
      v4f a = *(const v4f*)&src[i0];
      v4f c = *(const v4f*)&src[i0 + 4];
      v8h h;
#pragma unroll
      for (int e = 0; e < 4; ++e) { h[e] = (_Float16)a[e]; h[e + 4] = (_Float16)c[e]; }
      *(v8h*)&sm.sA[i0] = h;
    }
  }
  __syncthreads();

  const uint32_t half_total = ((uint32_t)gridDim.x * (uint32_t)NPIX) >> 1;

  for (int s = 0; s < ns; ++s) {
    // step key: jax.random.split(key(42), ns)[s]
    uint32_t k0s = tf_bits(0u, 42u, 2u * (uint32_t)s,      (uint32_t)ns);
    uint32_t k1s = tf_bits(0u, 42u, 2u * (uint32_t)s + 1u, (uint32_t)ns);

    // ---- Phase 1: perception -> WMMA layer1 -> WMMA layer2 -> stochastic out (sA -> sB) ----
#pragma unroll 1
    for (int i = 0; i < TILES_PER_WAVE; ++i) {
      const int t   = wave * TILES_PER_WAVE + i;   // 16-pixel tile index
      const int R   = t >> 2;
      const int col = ((t & 3) << 4) + lrow;       // this lane's pixel column

      // Build perception directly in B-matrix layout: element e -> K = lgrp*16 + e
      // bA covers K=0..31 (x ch0-15, sobelX ch0-15); bB covers K=32..63 (sobelY ch0-15, zero pad)
      v16h bA, bB;
      if (lgrp == 0) {
#pragma unroll
        for (int c = 0; c < 16; ++c)
          bA[c] = sm.sA[c * NPIX + R * WWD + col];
#pragma unroll
        for (int c = 0; c < 16; ++c) {
          float t0 = ldz(sm.sA, c, R - 1, col - 1), t1 = ldz(sm.sA, c, R - 1, col), t2 = ldz(sm.sA, c, R - 1, col + 1);
          float u0 = ldz(sm.sA, c, R + 1, col - 1), u1 = ldz(sm.sA, c, R + 1, col), u2 = ldz(sm.sA, c, R + 1, col + 1);
          bB[c] = (_Float16)((t0 + 2.0f * t1 + t2) - (u0 + 2.0f * u1 + u2));   // sobelY
        }
      } else {
#pragma unroll
        for (int c = 0; c < 16; ++c) {
          float l0 = ldz(sm.sA, c, R - 1, col - 1), l1 = ldz(sm.sA, c, R, col - 1), l2 = ldz(sm.sA, c, R + 1, col - 1);
          float r0 = ldz(sm.sA, c, R - 1, col + 1), r1 = ldz(sm.sA, c, R, col + 1), r2 = ldz(sm.sA, c, R + 1, col + 1);
          bA[c] = (_Float16)((l0 - r0) + 2.0f * (l1 - r1) + (l2 - r2));        // sobelX
        }
#pragma unroll
        for (int c = 0; c < 16; ++c) bB[c] = (_Float16)0.0f;
      }

      // Layer 1: h = relu(W1 * perception + b1). 8 live accumulators, WMMAs issued in two
      // interleaved passes so C-operand dependencies are 8 WMMAs apart (hides XDL latency
      // and the WMMA->VALU hazard NOPs behind other tiles' WMMAs).
      v8f acc[8];
#pragma unroll
      for (int m = 0; m < 8; ++m) acc[m] = bias1[m];
#pragma unroll
      for (int m = 0; m < 8; ++m)
        acc[m] = __builtin_amdgcn_wmma_f32_16x16x32_f16(false, A1[2 * m],     false, bA, (short)0, acc[m], false, false);
#pragma unroll
      for (int m = 0; m < 8; ++m)
        acc[m] = __builtin_amdgcn_wmma_f32_16x16x32_f16(false, A1[2 * m + 1], false, bB, (short)0, acc[m], false, false);
#pragma unroll
      for (int m = 0; m < 8; ++m) {
        v8h hv;   // relu + f16, stored in B-ready [pixel][K] layout (one 16B store per lane)
#pragma unroll
        for (int r = 0; r < 8; ++r) hv[r] = (_Float16)fmaxf(acc[m][r], 0.0f);
        *(v8h*)&sm.hscr[wave][lrow][m * 16 + lgrp * 8] = hv;
      }

      // Layer 2: diff = W2 * h + b2, split into two accumulator chains
      v8f d0 = bias2, d1;
#pragma unroll
      for (int r = 0; r < 8; ++r) d1[r] = 0.0f;
      {
        v16h bj0 = *(const v16h*)&sm.hscr[wave][lrow][0 * 32 + lgrp * 16];
        v16h bj1 = *(const v16h*)&sm.hscr[wave][lrow][1 * 32 + lgrp * 16];
        v16h bj2 = *(const v16h*)&sm.hscr[wave][lrow][2 * 32 + lgrp * 16];
        v16h bj3 = *(const v16h*)&sm.hscr[wave][lrow][3 * 32 + lgrp * 16];
        d0 = __builtin_amdgcn_wmma_f32_16x16x32_f16(false, A2[0], false, bj0, (short)0, d0, false, false);
        d1 = __builtin_amdgcn_wmma_f32_16x16x32_f16(false, A2[2], false, bj2, (short)0, d1, false, false);
        d0 = __builtin_amdgcn_wmma_f32_16x16x32_f16(false, A2[1], false, bj1, (short)0, d0, false, false);
        d1 = __builtin_amdgcn_wmma_f32_16x16x32_f16(false, A2[3], false, bj3, (short)0, d1, false, false);
      }

      // Stochastic mask (jax.random.uniform(key_s, (B,1,H,W)) < 0.5) and out = x + diff*stoch
      uint32_t pidx = (uint32_t)b * (uint32_t)NPIX + (uint32_t)(R * WWD + col);
      uint32_t rb = tf_bits(k0s, k1s, pidx, half_total);
      union { uint32_t u; float f; } cv; cv.u = (rb >> 9) | 0x3f800000u;
      float stoch = ((cv.f - 1.0f) < 0.5f) ? 1.0f : 0.0f;
#pragma unroll
      for (int r = 0; r < 8; ++r) {
        int ch = lgrp * 8 + r;                         // D layout: M = lgrp*8 + r
        float xv = (float)sm.sA[ch * NPIX + R * WWD + col];
        sm.sB[ch * NPIX + R * WWD + col] = (_Float16)(xv + (d0[r] + d1[r]) * stoch);
      }
    }
    __syncthreads();

    // ---- Phase 2: 3x3 max-pool alive gate on alpha (ch 3) + clip (sB -> sA), vectorized ----
    {
      const int p0   = tid * 16;       // 16 contiguous pixels in one row, 32B aligned
      const int prow = p0 >> 6;
      const int c0   = p0 & 63;

      float cm[18];                    // per-column max over the 3 rows
#pragma unroll
      for (int i = 0; i < 18; ++i) cm[i] = -3.0e38f;
#pragma unroll
      for (int dr = -1; dr <= 1; ++dr) {
        int rr = prow + dr;
        if ((unsigned)rr < (unsigned)HH) {
          v16h a = *(const v16h*)&sm.sB[3 * NPIX + rr * WWD + c0];
#pragma unroll
          for (int i = 0; i < 16; ++i) cm[i + 1] = fmaxf(cm[i + 1], (float)a[i]);
          if (c0 > 0)  cm[0]  = fmaxf(cm[0],  (float)sm.sB[3 * NPIX + rr * WWD + c0 - 1]);
          if (c0 < 48) cm[17] = fmaxf(cm[17], (float)sm.sB[3 * NPIX + rr * WWD + c0 + 16]);
        }
      }
      float alive[16];
#pragma unroll
      for (int i = 0; i < 16; ++i) {
        float mx = fmaxf(fmaxf(cm[i], cm[i + 1]), cm[i + 2]);
        alive[i] = (mx > 0.1f) ? 1.0f : 0.0f;
      }
#pragma unroll
      for (int ch = 0; ch < NCH; ++ch) {
        v16h vv = *(const v16h*)&sm.sB[ch * NPIX + p0];
        v16h ov;
#pragma unroll
        for (int i = 0; i < 16; ++i) {
          float v = (float)vv[i] * alive[i];
          ov[i] = (_Float16)fminf(fmaxf(v, 0.0f), 1.0f);
        }
        *(v16h*)&sm.sA[ch * NPIX + p0] = ov;
      }
    }
    __syncthreads();
  }

  // ---- Write final state back to HBM as f32 ----
  {
    float* dst = out + (size_t)b * NCH * NPIX;
#pragma unroll 1
    for (int i0 = tid * 256; i0 < tid * 256 + 256; i0 += 8) {
      v8h h = *(const v8h*)&sm.sA[i0];
      v4f a, c;
#pragma unroll
      for (int e = 0; e < 4; ++e) { a[e] = (float)h[e]; c[e] = (float)h[e + 4]; }
      *(v4f*)&dst[i0]     = a;
      *(v4f*)&dst[i0 + 4] = c;
    }
  }
}

extern "C" void kernel_launch(void* const* d_in, const int* in_sizes, int n_in,
                              void* d_out, int out_size, void* d_ws, size_t ws_size,
                              hipStream_t stream) {
  (void)n_in; (void)out_size; (void)d_ws; (void)ws_size;
  const float* inp = (const float*)d_in[0];
  const float* W1  = (const float*)d_in[1];
  const float* b1  = (const float*)d_in[2];
  const float* W2  = (const float*)d_in[3];
  const float* b2  = (const float*)d_in[4];
  const int* nst   = (const int*)d_in[5];
  const int B = in_sizes[0] / (NCH * NPIX);   // 8 images, one workgroup (one WGP) each
  nca_persistent_kernel<<<dim3(B), dim3(NTHREADS), 0, stream>>>(inp, W1, b1, W2, b2, nst, (float*)d_out);
}
